// Model_HyperDrop_OGB_61838939128325
// MI455X (gfx1250) — matrix operand order repaired and checked by reading, not compiled, not run on hardware
//
#include <hip/hip_runtime.h>
#include <hip/hip_bf16.h>

typedef __attribute__((ext_vector_type(16))) _Float16 v16h;
typedef __attribute__((ext_vector_type(8)))  float    v8f;

#define NT     65536      // N_TOTAL
#define DCH    128        // NHID
#define BGR    256        // B_GRAPHS
#define NPG    256        // N_PER_G
#define E0     131072
#define EG0    512
#define KEEP0  410        // ceil(0.8*512)
#define E1     104960     // 256*410
#define EG1    410
#define KEEP1  328        // ceil(0.8*410)
#define E2     83968      // 256*328

// ---------------------------------------------------------------------------
// Generic WMMA GEMM: C[M,N] = act(A[M,K] @ B[K,N] + bias[N])
// A,B,C fp32 row-major; A/B converted to f16 in-register.
// K is a compile-time template parameter -> K-loop fully unrolls into
// back-to-back v_wmma_f32_16x16x32_f16 ops (4 for K=128, 12 for K=384).
// One wave -> one 16x16 tile. block=128 (4 waves), grid=(N/16, M/64).
// M % 64 == 0, N % 16 == 0, K % 32 == 0 (guaranteed by problem sizes).
// act: 0=none, 1=relu
// ---------------------------------------------------------------------------
template <int K>
__global__ void gemm_wmma_f16(const float* __restrict__ A,
                              const float* __restrict__ Bm,
                              const float* __restrict__ bias,
                              float* __restrict__ C,
                              int M, int N, int act) {
    const int lane = threadIdx.x & 31;
    const int wave = threadIdx.x >> 5;
    const int n0 = blockIdx.x * 16;
    const int m0 = (blockIdx.y * 4 + wave) * 16;
    const int l16 = lane & 15;
    const int mrow = m0 + l16;
    const int ncol = n0 + l16;
    const int kbA = (lane < 16) ? 0 : 16;   // A half-wave K base
    const int kbB = (lane < 16) ? 0 : 16;   // B half-wave K base

    v8f acc = {};
    const float* Arow = A + (size_t)mrow * K;
#pragma unroll
    for (int k0 = 0; k0 < K; k0 += 32) {
        v16h a, b;
#pragma unroll
        for (int i = 0; i < 8; ++i) {
            a[i]     = (_Float16)Arow[k0 + kbA + i];
            a[i + 8] = (_Float16)Arow[k0 + kbA + 8 + i];
        }
#pragma unroll
        for (int i = 0; i < 16; ++i) {
            b[i] = (_Float16)Bm[(size_t)(k0 + kbB + i) * N + ncol];
        }
        acc = __builtin_amdgcn_wmma_f32_16x16x32_f16(
            false, a, false, b, (short)0, acc, false, false);
    }
    const float bv = bias ? bias[ncol] : 0.0f;
    const int rsel = (lane < 16) ? 0 : 8;
#pragma unroll
    for (int r = 0; r < 8; ++r) {
        float v = acc[r] + bv;
        if (act == 1) v = fmaxf(v, 0.0f);
        C[(size_t)(m0 + r + rsel) * N + ncol] = v;
    }
}

// ---------------------------------------------------------------------------
// Elementwise / embedding / scatter kernels
// ---------------------------------------------------------------------------
__global__ void zero_f32(float* __restrict__ p, int n) {
    int t = blockIdx.x * blockDim.x + threadIdx.x;
    if (t < n) p[t] = 0.0f;
}

__global__ void embed_sum(const int* __restrict__ idx, const float* __restrict__ table,
                          float* __restrict__ out, int rows, int nidx) {
    int t = blockIdx.x * blockDim.x + threadIdx.x;   // rows*128 threads
    if (t >= rows * DCH) return;
    int r = t >> 7, c = t & 127;
    float acc = 0.0f;
    for (int j = 0; j < nidx; ++j)
        acc += table[(size_t)idx[r * nidx + j] * DCH + c];
    out[t] = acc;
}

__global__ void init_deg_bc(float* __restrict__ deg, float* __restrict__ bc) {
    int t = blockIdx.x * blockDim.x + threadIdx.x;
    if (t < NT) { deg[t] = 1.0f; bc[t] = 0.0f; }
}

__global__ void edge_deg_bc(const int* __restrict__ src, const int* __restrict__ dst,
                            float* __restrict__ deg, float* __restrict__ bc, int E) {
    int t = blockIdx.x * blockDim.x + threadIdx.x;
    if (t >= E) return;
    int s = src[t], d = dst[t];
    atomicAdd(deg + s, 1.0f);
    atomicAdd(bc + s, 1.0f);
    atomicAdd(bc + d, 1.0f);
}

// x_new[n,c] = relu(h[n,c] + root[c]) / deg[n]   (initializes the accumulator)
__global__ void self_term(const float* __restrict__ h, const float* __restrict__ root,
                          const float* __restrict__ deg, float* __restrict__ xnew) {
    int t = blockIdx.x * blockDim.x + threadIdx.x;
    if (t >= NT * DCH) return;
    int n = t >> 7, c = t & 127;
    xnew[t] = fmaxf(h[t] + root[c], 0.0f) / deg[n];
}

// scatter: x_new[col] += dis[row]*dis[col]*relu(h[row]+e)*[ew]
__global__ void message_pass(const float* __restrict__ h, const float* __restrict__ e,
                             const int* __restrict__ src, const int* __restrict__ dst,
                             const float* __restrict__ deg, const float* __restrict__ ew,
                             float* __restrict__ xnew, int E) {
    int t = blockIdx.x * blockDim.x + threadIdx.x;
    if (t >= E * DCH) return;
    int ei = t >> 7, c = t & 127;
    int r = src[ei], co = dst[ei];
    float v = __frsqrt_rn(deg[r]) * __frsqrt_rn(deg[co]) *
              fmaxf(h[(size_t)r * DCH + c] + e[t], 0.0f);
    if (ew) v *= ew[ei];
    atomicAdd(xnew + (size_t)co * DCH + c, v);
}

__global__ void relu_copy(const float* __restrict__ a, float* __restrict__ b, int n) {
    int t = blockIdx.x * blockDim.x + threadIdx.x;
    if (t < n) b[t] = fmaxf(a[t], 0.0f);
}

// xs[g, 0:128]+=max, [128:256]+=mean, [256:384]+=sum over nodes of graph g
__global__ void readout(const float* __restrict__ x, float* __restrict__ xs) {
    int g = blockIdx.x, c = threadIdx.x;        // 256 blocks x 128 threads
    const float* xp = x + (size_t)g * NPG * DCH + c;
    float mx = -1e30f, sm = 0.0f;
    for (int n = 0; n < NPG; ++n) {
        float v = xp[(size_t)n * DCH];
        mx = fmaxf(mx, v);
        sm += v;
    }
    float* o = xs + (size_t)g * (3 * DCH);
    o[c]            += mx;
    o[DCH + c]      += sm * (1.0f / NPG);
    o[2 * DCH + c]  += sm;
}

// ---------------------------------------------------------------------------
// Hypergraph conv pieces (generic over dout = 128 or 1)
// ---------------------------------------------------------------------------
__global__ void hyper_scatter(const float* __restrict__ z, float* __restrict__ m,
                              const float* __restrict__ bc,
                              const int* __restrict__ src, const int* __restrict__ dst,
                              int E, int dout) {
    int t = blockIdx.x * blockDim.x + threadIdx.x;
    if (t >= E * dout) return;
    int ei = t / dout, c = t - ei * dout;
    float v = z[t];
    int s = src[ei], d = dst[ei];
    if (bc[s] != 1.0f) atomicAdd(m + (size_t)s * dout + c, v);
    if (bc[d] != 1.0f) atomicAdd(m + (size_t)d * dout + c, v);
}

__global__ void hyper_scale(float* __restrict__ m, const float* __restrict__ bc, int dout) {
    int t = blockIdx.x * blockDim.x + threadIdx.x;
    if (t >= NT * dout) return;
    int n = t / dout;
    float b = bc[n];
    float inv = (b != 1.0f && b > 0.0f) ? 1.0f / b : 0.0f;
    m[t] *= inv;
}

// out = (w_s*m[src] + w_d*m[dst] + z) / Dn + bias ; act 1=relu 2=tanh
__global__ void hyper_gather(const float* __restrict__ z, const float* __restrict__ m,
                             const float* __restrict__ bc,
                             const int* __restrict__ src, const int* __restrict__ dst,
                             const float* __restrict__ bias, float* __restrict__ out,
                             int E, int dout, int act) {
    int t = blockIdx.x * blockDim.x + threadIdx.x;
    if (t >= E * dout) return;
    int ei = t / dout, c = t - ei * dout;
    int s = src[ei], d = dst[ei];
    bool ws = (bc[s] != 1.0f), wd = (bc[d] != 1.0f);
    float Dn = (ws ? 1.0f : 0.0f) + (wd ? 1.0f : 0.0f) + 1.0f;
    float v = (ws ? m[(size_t)s * dout + c] : 0.0f)
            + (wd ? m[(size_t)d * dout + c] : 0.0f)
            + z[t];
    v = v / Dn + bias[c];
    if (act == 1) v = fmaxf(v, 0.0f);
    else if (act == 2) v = tanhf(v);
    out[t] = v;
}

// z_sc[ei] = dot(e[ei,:], w[0:128])
__global__ void matvec128(const float* __restrict__ feat, const float* __restrict__ w,
                          float* __restrict__ out, int E) {
    int ei = blockIdx.x * blockDim.x + threadIdx.x;
    if (ei >= E) return;
    const float4* f = (const float4*)(feat + (size_t)ei * DCH);
    const float4* wv = (const float4*)w;
    float acc = 0.0f;
#pragma unroll 8
    for (int i = 0; i < 32; ++i) {
        float4 a = f[i], b = wv[i];
        acc += a.x * b.x + a.y * b.y + a.z * b.z + a.w * b.w;
    }
    out[ei] = acc;
}

// ---------------------------------------------------------------------------
// Per-graph top-k via bitonic sort of 512 (score,idx) pairs in LDS.
// block = 256 threads, one graph per block.
// ---------------------------------------------------------------------------
__global__ void topk_select(const float* __restrict__ score,
                            const int* __restrict__ src, const int* __restrict__ dst,
                            int Eg, int keep,
                            int* __restrict__ perm, int* __restrict__ nsrc,
                            int* __restrict__ ndst, float* __restrict__ new_ew) {
    __shared__ float key[512];
    __shared__ int   sidx[512];
    int g = blockIdx.x, t = threadIdx.x;
    for (int i = t; i < 512; i += 256) {
        key[i]  = (i < Eg) ? score[(size_t)g * Eg + i] : -1e30f;
        sidx[i] = i;
    }
    __syncthreads();
    for (int ksz = 2; ksz <= 512; ksz <<= 1) {
        for (int j = ksz >> 1; j > 0; j >>= 1) {
            for (int i = t; i < 512; i += 256) {
                int p = i ^ j;
                if (p > i) {
                    bool desc = ((i & ksz) == 0);
                    float a = key[i], b = key[p];
                    bool sw = desc ? (a < b) : (a > b);
                    if (sw) {
                        key[i] = b; key[p] = a;
                        int tmp = sidx[i]; sidx[i] = sidx[p]; sidx[p] = tmp;
                    }
                }
            }
            __syncthreads();
        }
    }
    for (int j = t; j < keep; j += 256) {
        int oe = g * Eg + sidx[j];
        int ne = g * keep + j;
        perm[ne]   = oe;
        nsrc[ne]   = src[oe];
        ndst[ne]   = dst[oe];
        new_ew[ne] = fminf(fmaxf(key[j], 0.0f), 1.0f);
    }
}

__global__ void permute_rows(const float* __restrict__ in, const int* __restrict__ perm,
                             float* __restrict__ out, int nkeep) {
    int t = blockIdx.x * blockDim.x + threadIdx.x;
    if (t >= nkeep * DCH) return;
    int j = t >> 7, c = t & 127;
    out[t] = in[(size_t)perm[j] * DCH + c];
}

// ---------------------------------------------------------------------------
// Host side
// ---------------------------------------------------------------------------
template <int K>
static inline void launch_gemm(const float* A, const float* Bm, const float* bias,
                               float* C, int M, int N, int act, hipStream_t s) {
    dim3 grid(N / 16, M / 64);
    gemm_wmma_f16<K><<<grid, 128, 0, s>>>(A, Bm, bias, C, M, N, act);
}

extern "C" void kernel_launch(void* const* d_in, const int* in_sizes, int n_in,
                              void* d_out, int out_size, void* d_ws, size_t ws_size,
                              hipStream_t stream) {
    const int*   x_idx      = (const int*)  d_in[0];   // NT x 9
    const int*   ebond_idx  = (const int*)  d_in[1];   // E0 x 3
    const int*   edge_index = (const int*)  d_in[2];   // 2 x E0
    const float* atom_table = (const float*)d_in[4];
    const float* bond_table = (const float*)d_in[5];
    const float* W_gcn      = (const float*)d_in[6];   // 3 x 128 x 128
    const float* b_gcn      = (const float*)d_in[7];   // 3 x 128
    const float* root_emb   = (const float*)d_in[8];   // 3 x 128
    const float* W_hyp      = (const float*)d_in[9];   // 2 x 128 x 128
    const float* b_hyp      = (const float*)d_in[10];  // 2 x 128
    const float* W_sc       = (const float*)d_in[11];  // 2 x 128 x 1
    const float* b_sc       = (const float*)d_in[12];  // 2 x 1
    const float* Wc1        = (const float*)d_in[13];  // 384 x 128
    const float* bc1        = (const float*)d_in[14];
    const float* Wc2        = (const float*)d_in[15];  // 128 x 64
    const float* bc2        = (const float*)d_in[16];
    const float* Wc3        = (const float*)d_in[17];  // 64 x 128
    const float* bc3        = (const float*)d_in[18];

    // ---- workspace layout (floats) ----
    float* x      = (float*)d_ws;                 // NT*128
    float* h      = x      + (size_t)NT * DCH;    // NT*128
    float* xnew   = h      + (size_t)NT * DCH;    // NT*128
    float* z      = h;                            // alias over h+xnew (E0*128), dead by then
    float* e      = xnew   + (size_t)NT * DCH;    // E0*128
    float* e2     = e      + (size_t)E0 * DCH;    // E0*128
    float* m_node = e2     + (size_t)E0 * DCH;    // NT*128
    float* deg    = m_node + (size_t)NT * DCH;    // NT
    float* bc     = deg    + NT;                  // NT
    float* score  = bc     + NT;                  // E0
    float* zsc    = score  + E0;                  // E0
    float* m_s    = zsc    + E0;                  // NT
    float* ew1    = m_s    + NT;                  // E0
    float* ew2    = ew1    + E0;                  // E0
    float* xs     = ew2    + E0;                  // BGR*384
    float* h1     = xs     + BGR * 3 * DCH;       // 256*128
    float* h2     = h1     + BGR * DCH;           // 256*64
    int*   src1   = (int*)(h2 + BGR * 64);        // E0 ints
    int*   dst1   = src1 + E0;
    int*   src2   = dst1 + E0;
    int*   dst2   = src2 + E0;
    int*   permb  = dst2 + E0;

    const int TB = 256;

    // embeddings: x = sum atom_table rows, e = sum bond_table rows
    embed_sum<<<(NT * DCH) / TB, TB, 0, stream>>>(x_idx, atom_table, x, NT, 9);
    embed_sum<<<(E0 * DCH) / TB, TB, 0, stream>>>(ebond_idx, bond_table, e, E0, 3);
    zero_f32<<<(BGR * 3 * DCH + TB - 1) / TB, TB, 0, stream>>>(xs, BGR * 3 * DCH);

    const int Es[3]   = {E0, E1, E2};
    const int Egs[3]  = {EG0, EG1, 0};
    const int kps[2]  = {KEEP0, KEEP1};
    const int* cs     = edge_index;          // layer-0 src
    const int* cd     = edge_index + E0;     // layer-0 dst
    const float* cew  = nullptr;
    int* outS[2] = {src1, src2};
    int* outD[2] = {dst1, dst2};
    float* outW[2] = {ew1, ew2};

    for (int l = 0; l < 3; ++l) {
        const int E = Es[l];

        // degrees + incidence counts
        init_deg_bc<<<NT / TB, TB, 0, stream>>>(deg, bc);
        edge_deg_bc<<<E / TB, TB, 0, stream>>>(cs, cd, deg, bc, E);

        // h = x @ W_gcn[l] + b_gcn[l]          (WMMA, K=128 -> 4 unrolled wmma)
        launch_gemm<DCH>(x, W_gcn + (size_t)l * DCH * DCH, b_gcn + (size_t)l * DCH,
                         h, NT, DCH, 0, stream);

        // self term initializes accumulator, then edge messages scatter in
        self_term<<<(NT * DCH) / TB, TB, 0, stream>>>(h, root_emb + (size_t)l * DCH, deg, xnew);
        message_pass<<<(E * DCH) / TB, TB, 0, stream>>>(h, e, cs, cd, deg, cew, xnew, E);
        relu_copy<<<(NT * DCH) / TB, TB, 0, stream>>>(xnew, x, NT * DCH);

        // readout accumulation (x is final for this layer)
        readout<<<BGR, DCH, 0, stream>>>(x, xs);

        if (l < 2) {
            // ---- hypergraph conv on e (dout=128) ----
            launch_gemm<DCH>(e, W_hyp + (size_t)l * DCH * DCH, nullptr, z, E, DCH, 0, stream);
            zero_f32<<<(NT * DCH) / TB, TB, 0, stream>>>(m_node, NT * DCH);
            hyper_scatter<<<(E * DCH) / TB, TB, 0, stream>>>(z, m_node, bc, cs, cd, E, DCH);
            hyper_scale<<<(NT * DCH) / TB, TB, 0, stream>>>(m_node, bc, DCH);
            hyper_gather<<<(E * DCH) / TB, TB, 0, stream>>>(z, m_node, bc, cs, cd,
                                                           b_hyp + (size_t)l * DCH, e2,
                                                           E, DCH, /*relu*/1);
            // ---- score = tanh(hconv(e2, W_sc, b_sc)) (dout=1) ----
            matvec128<<<(E + TB - 1) / TB, TB, 0, stream>>>(e2, W_sc + (size_t)l * DCH, zsc, E);
            zero_f32<<<(NT + TB - 1) / TB, TB, 0, stream>>>(m_s, NT);
            hyper_scatter<<<(E + TB - 1) / TB, TB, 0, stream>>>(zsc, m_s, bc, cs, cd, E, 1);
            hyper_scale<<<(NT + TB - 1) / TB, TB, 0, stream>>>(m_s, bc, 1);
            hyper_gather<<<(E + TB - 1) / TB, TB, 0, stream>>>(zsc, m_s, bc, cs, cd,
                                                              b_sc + l, score, E, 1, /*tanh*/2);
            // ---- per-graph top-k ----
            const int Eg = Egs[l], keep = kps[l];
            topk_select<<<BGR, 256, 0, stream>>>(score, cs, cd, Eg, keep,
                                                 permb, outS[l], outD[l], outW[l]);
            const int nkeep = BGR * keep;
            permute_rows<<<(nkeep * DCH) / TB, TB, 0, stream>>>(e2, permb, e, nkeep);

            cs = outS[l]; cd = outD[l]; cew = outW[l];
        }
    }

    // MLP head (WMMA): 256x384x128 -> 256x128x64 -> 256x64x128
    launch_gemm<3 * DCH>(xs, Wc1, bc1, h1, BGR, DCH, 1, stream);
    launch_gemm<DCH>(h1, Wc2, bc2, h2, BGR, 64, 1, stream);
    launch_gemm<64>(h2, Wc3, bc3, (float*)d_out, BGR, DCH, 0, stream);
}
